// SDT_35064113004822
// MI455X (gfx1250) — matrix operand order repaired and checked by reading, not compiled, not run on hardware
//
#include <hip/hip_runtime.h>

typedef __attribute__((ext_vector_type(16))) __bf16 v16bf;
typedef __attribute__((ext_vector_type(8)))  float  v8f;

#define B_ROWS 8192
#define D_DIM  512
#define NN     255
#define NPAD   256
#define LEAVES 256
#define O_DIM  64

// ---------------------------------------------------------------------------
// Prep kernel A: pack Ws (512x255 f32, row-major) into bf16 WMMA B-fragments,
// padded to 256 columns, and build padded bias.
// Layout: p1[((tileN*16 + kb)*32 + lane)*16 + i]
//   colB = tileN*16 + (lane&15); k = kb*32 + (lane>>4)*8 + (i&7) + (i>>3)*16
// => each lane's fragment for (tileN, kb) is one contiguous 32B chunk.
// ---------------------------------------------------------------------------
__global__ __launch_bounds__(256) void pack_ws(
    const float* __restrict__ Ws, const float* __restrict__ bs,
    __bf16* __restrict__ p1, float* __restrict__ bs_pad)
{
  const int tid  = blockIdx.x * 256 + threadIdx.x;   // 131072 total
  const int i    = tid & 15;
  const int lane = (tid >> 4) & 31;
  const int kb   = (tid >> 9) & 15;
  const int tn   = tid >> 13;
  const int col  = tn * 16 + (lane & 15);
  const int k    = kb * 32 + (lane >> 4) * 8 + (i & 7) + (i >> 3) * 16;
  const float v  = (col < NN) ? Ws[(size_t)k * NN + col] : 0.0f;
  p1[tid] = (__bf16)v;
  if (tid < NPAD) bs_pad[tid] = (tid < NN) ? bs[tid] : 0.0f;
}

// Prep kernel B: pack value (256x64 f32) into bf16 WMMA B-fragments.
// p2[((w*8 + kb)*32 + lane)*16 + i]; col = w*16+(lane&15); k = kb*32+(lane>>4)*8+(i&7)+(i>>3)*16
__global__ __launch_bounds__(256) void pack_value(
    const float* __restrict__ value, __bf16* __restrict__ p2)
{
  const int tid  = blockIdx.x * 256 + threadIdx.x;   // 16384 total
  const int i    = tid & 15;
  const int lane = (tid >> 4) & 31;
  const int kb   = (tid >> 9) & 7;
  const int w    = tid >> 12;
  const int col  = w * 16 + (lane & 15);
  const int k    = kb * 32 + (lane >> 4) * 8 + (i & 7) + (i >> 3) * 16;
  p2[tid] = (__bf16)value[(size_t)k * O_DIM + col];
}

// ---------------------------------------------------------------------------
// Kernel 1: probs = sigmoid(x @ Ws + bs) + regularizer partials.
// One wave per (16-row M-tile, group of 4 N-tiles): 2048 waves.
// Inner loop: 1 A-fragment (f32 loads + cvt_pk_bf16) feeds 4 WMMAs.
// ---------------------------------------------------------------------------
__global__ __launch_bounds__(256) void sdt_gemm_sigmoid(
    const float* __restrict__ x, const __bf16* __restrict__ p1,
    const float* __restrict__ bs_pad, float* __restrict__ probs,
    float* __restrict__ partials)
{
  __shared__ float red[256];
  const int lane  = threadIdx.x & 31;
  const int wave  = threadIdx.x >> 5;
  const int id    = blockIdx.x * 8 + wave;   // 2048 wave-jobs
  const int tileM = id >> 2;                 // 0..511
  const int ng    = id & 3;                  // N-group: tiles ng*4 .. ng*4+3
  const int half  = lane >> 4;
  const int lm    = lane & 15;
  const int rowA  = tileM * 16 + lm;

  v8f acc[4] = {};
  const float*  ax = x  + (size_t)rowA * D_DIM + half * 8;
  const __bf16* bp = p1 + ((size_t)(ng * 4) * 16 * 32 + lane) * 16;

  for (int kb = 0; kb < 16; ++kb) {
    const float* a = ax + kb * 32;
    float4 a0 = *(const float4*)(a);
    float4 a1 = *(const float4*)(a + 4);
    float4 a2 = *(const float4*)(a + 16);
    float4 a3 = *(const float4*)(a + 20);
    v16bf A;
    A[0]=(__bf16)a0.x;  A[1]=(__bf16)a0.y;  A[2]=(__bf16)a0.z;  A[3]=(__bf16)a0.w;
    A[4]=(__bf16)a1.x;  A[5]=(__bf16)a1.y;  A[6]=(__bf16)a1.z;  A[7]=(__bf16)a1.w;
    A[8]=(__bf16)a2.x;  A[9]=(__bf16)a2.y;  A[10]=(__bf16)a2.z; A[11]=(__bf16)a2.w;
    A[12]=(__bf16)a3.x; A[13]=(__bf16)a3.y; A[14]=(__bf16)a3.z; A[15]=(__bf16)a3.w;

    v16bf B0 = *(const v16bf*)(bp + (size_t)(0 * 16 + kb) * 512);
    v16bf B1 = *(const v16bf*)(bp + (size_t)(1 * 16 + kb) * 512);
    v16bf B2 = *(const v16bf*)(bp + (size_t)(2 * 16 + kb) * 512);
    v16bf B3 = *(const v16bf*)(bp + (size_t)(3 * 16 + kb) * 512);

    acc[0] = __builtin_amdgcn_wmma_f32_16x16x32_bf16(false, A, false, B0, (short)0, acc[0], false, false);
    acc[1] = __builtin_amdgcn_wmma_f32_16x16x32_bf16(false, A, false, B1, (short)0, acc[1], false, false);
    acc[2] = __builtin_amdgcn_wmma_f32_16x16x32_bf16(false, A, false, B2, (short)0, acc[2], false, false);
    acc[3] = __builtin_amdgcn_wmma_f32_16x16x32_bf16(false, A, false, B3, (short)0, acc[3], false, false);
  }

  float lsum = 0.0f;
  #pragma unroll
  for (int nt = 0; nt < 4; ++nt) {
    const int colB = (ng * 4 + nt) * 16 + lm;
    const float bias = bs_pad[colB];
    const int dd = 31 - __clz(colB + 1);
    const float wreg = (colB < NN)
        ? -0.5f * exp2f(-(float)dd) * (1.0f / (float)B_ROWS) : 0.0f;
    #pragma unroll
    for (int r = 0; r < 8; ++r) {
      const int m = r + half * 8;
      const float z = acc[nt][r] + bias;
      const float p = 1.0f / (1.0f + __expf(-z));
      probs[(size_t)(tileM * 16 + m) * NPAD + colB] = p;
      lsum += wreg * __logf(fmaxf(p * (1.0f - p), 1e-5f));
    }
  }

  red[threadIdx.x] = lsum;
  __syncthreads();
  for (int s = 128; s > 0; s >>= 1) {
    if (threadIdx.x < s) red[threadIdx.x] += red[threadIdx.x + s];
    __syncthreads();
  }
  if (threadIdx.x == 0) partials[blockIdx.x] = red[0];
}

// ---------------------------------------------------------------------------
// Kernel 2: per 16-row block: stage probs in LDS (coalesced), compute leaf
// path products, store bf16 directly in WMMA A-fragment layout in LDS, then
// out = P(16x256) @ value(256x64): 4 waves x 8 WMMAs (A via 32B ds loads).
// ---------------------------------------------------------------------------
__global__ __launch_bounds__(128) void sdt_leaf_gemm(
    const float* __restrict__ probs, const __bf16* __restrict__ p2,
    float* __restrict__ out)
{
  __shared__ float Ps[16][NPAD];                                  // 16 KB
  __shared__ __attribute__((aligned(32))) __bf16 Abf[8 * 32 * 16]; // 8 KB
  const int rowBase = blockIdx.x * 16;

  // Stage probs block (16x256 f32) with float4 loads.
  for (int it = 0; it < 8; ++it) {
    const int e = (threadIdx.x + it * 128) * 4;
    const int r = e >> 8, c0 = e & 255;
    *(float4*)&Ps[r][c0] =
        *(const float4*)(probs + (size_t)(rowBase + r) * NPAD + c0);
  }
  __syncthreads();

  // Leaf products -> bf16 A-fragments.
  for (int it = 0; it < 32; ++it) {
    const int e = threadIdx.x + it * 128;     // 4096 (r,l) pairs
    const int r = e >> 8, l = e & 255;
    float a = 1.0f;
    #pragma unroll
    for (int d = 0; d < 8; ++d) {
      const int node = ((1 << d) - 1) + (l >> (8 - d));
      const int bit  = (l >> (7 - d)) & 1;
      const float p  = Ps[r][node];
      const float f  = bit ? (1.0f - p) : p;
      a *= fmaxf(f, 1e-5f);
    }
    // leaf l = kb*32 + w ; w -> (half, group, j) per 16-bit A layout
    const int kb = l >> 5, w = l & 31;
    const int hf = (w >> 3) & 1, grp = w >> 4, j = w & 7;
    Abf[(size_t)(kb * 32 + hf * 16 + r) * 16 + grp * 8 + j] = (__bf16)a;
  }
  __syncthreads();

  const int wave = threadIdx.x >> 5;
  const int lane = threadIdx.x & 31;
  const int half = lane >> 4;
  const int lm   = lane & 15;
  const __bf16* bp = p2 + ((size_t)(wave * 8) * 32 + lane) * 16;

  v8f c = {};
  #pragma unroll
  for (int kb = 0; kb < 8; ++kb) {
    v16bf A  = *(const v16bf*)&Abf[(size_t)(kb * 32 + lane) * 16];
    v16bf Bv = *(const v16bf*)(bp + (size_t)kb * 512);
    c = __builtin_amdgcn_wmma_f32_16x16x32_bf16(false, A, false, Bv,
                                                (short)0, c, false, false);
  }
  const int colB = wave * 16 + lm;
  #pragma unroll
  for (int r = 0; r < 8; ++r)
    out[(size_t)(rowBase + r + half * 8) * O_DIM + colB] = c[r];
}

// ---------------------------------------------------------------------------
// Kernel 3: deterministic fixed-order reduction of 256 block partials -> reg.
// ---------------------------------------------------------------------------
__global__ __launch_bounds__(256) void sdt_reduce(
    const float* __restrict__ partials, float* __restrict__ regOut)
{
  __shared__ float red[256];
  red[threadIdx.x] = partials[threadIdx.x];
  __syncthreads();
  for (int st = 128; st > 0; st >>= 1) {
    if (threadIdx.x < st) red[threadIdx.x] += red[threadIdx.x + st];
    __syncthreads();
  }
  if (threadIdx.x == 0) regOut[0] = red[0];
}

extern "C" void kernel_launch(void* const* d_in, const int* in_sizes, int n_in,
                              void* d_out, int out_size, void* d_ws, size_t ws_size,
                              hipStream_t stream) {
  (void)in_sizes; (void)n_in; (void)out_size; (void)ws_size;
  const float* x     = (const float*)d_in[0];   // (8192, 512)
  const float* Ws    = (const float*)d_in[1];   // (512, 255)
  const float* bs    = (const float*)d_in[2];   // (255,)
  const float* value = (const float*)d_in[3];   // (256, 64)
  float* out = (float*)d_out;                   // 8192*64 floats + 1 (reg)

  char* wsb = (char*)d_ws;
  float*  probs    = (float*)(wsb);                        // 8 MB
  float*  partials = (float*)(wsb + 8388608);              // 1 KB
  float*  bs_pad   = (float*)(wsb + 8389632);              // 1 KB
  __bf16* p1       = (__bf16*)(wsb + 8390656);             // 256 KB packed Ws
  __bf16* p2       = (__bf16*)(wsb + 8652800);             // 32 KB packed value

  pack_ws        <<<512, 256, 0, stream>>>(Ws, bs, p1, bs_pad);
  pack_value     <<<64,  256, 0, stream>>>(value, p2);
  sdt_gemm_sigmoid<<<256, 256, 0, stream>>>(x, p1, bs_pad, probs, partials);
  sdt_leaf_gemm  <<<512, 128, 0, stream>>>(probs, p2, out);
  sdt_reduce     <<<1,   256, 0, stream>>>(partials, out + (size_t)B_ROWS * O_DIM);
}
